// MultiheadSelfAttention_1022202217238
// MI455X (gfx1250) — matrix operand order repaired and checked by reading, not compiled, not run on hardware
//
#include <hip/hip_runtime.h>
#include <hip/hip_fp16.h>

typedef _Float16 f16;
typedef _Float16 v4h  __attribute__((ext_vector_type(4)));
typedef _Float16 v8h  __attribute__((ext_vector_type(8)));
typedef _Float16 v16h __attribute__((ext_vector_type(16)));
typedef float    v8f  __attribute__((ext_vector_type(8)));
typedef int      v4i  __attribute__((ext_vector_type(4)));

static constexpr int B_ = 2, L_ = 2048, D_ = 2048, H_ = 16, C_ = 128;
static constexpr int M_ = B_ * L_;    // 4096 rows of x / y
static constexpr int N3 = 3 * D_;     // 6144 qkv outputs

// ---------------------------------------------------------------------------
// CDNA5 async global->LDS copy (ASYNCcnt-tracked), with sync fallback
// ---------------------------------------------------------------------------
#if defined(__HIP_DEVICE_COMPILE__) &&                                        \
    __has_builtin(__builtin_amdgcn_global_load_async_to_lds_b128) &&          \
    __has_builtin(__builtin_amdgcn_s_wait_asynccnt)
#define CDNA5_ASYNC 1
#else
#define CDNA5_ASYNC 0
#endif

__device__ __forceinline__ void cp16_to_lds(const f16* g, f16* l) {
#if CDNA5_ASYNC
  // global_load_async_to_lds_b128: per-lane 16B global -> LDS, no VGPR data.
  // Builtin params are int4 pointers in AS1 (global) / AS3 (LDS).
  typedef __attribute__((address_space(1))) v4i* gp_t;
  typedef __attribute__((address_space(3))) v4i* lp_t;
  __builtin_amdgcn_global_load_async_to_lds_b128(
      (gp_t)(uintptr_t)g, (lp_t)(uint32_t)(uintptr_t)l, 0, 0);
#else
  *(v8h*)l = *(const v8h*)g;
#endif
}

__device__ __forceinline__ void wait_async_lds() {
#if CDNA5_ASYNC
  __builtin_amdgcn_s_wait_asynccnt(0);
#endif
}

// ---------------------------------------------------------------------------
// helpers
// ---------------------------------------------------------------------------
__device__ __forceinline__ v8f zero_v8f() {
  v8f z;
#pragma unroll
  for (int r = 0; r < 8; ++r) z[r] = 0.0f;
  return z;
}

// Assemble a v16h A/B fragment from two 16-byte chunks (callers use contiguous k = 16*(lane/16) + s).
__device__ __forceinline__ v16h ld2x8(const f16* p0, const f16* p1) {
  v8h a = *(const v8h*)p0;
  v8h b = *(const v8h*)p1;
  return __builtin_shufflevector(a, b, 0, 1, 2, 3, 4, 5, 6, 7,
                                 8, 9, 10, 11, 12, 13, 14, 15);
}
__device__ __forceinline__ v16h ld32B(const f16* p) { return ld2x8(p, p + 8); }

__device__ __forceinline__ v8f wmma_f16(v16h a, v16h b, v8f c) {
  // D = A(16x32 f16) * B(32x16 f16) + C(16x16 f32)
  return __builtin_amdgcn_wmma_f32_16x16x32_f16(false, a, false, b,
                                                (short)0, c, false, false);
}

// ---------------------------------------------------------------------------
// Kernel 1: f32 -> f16 conversion (8 elements / thread, b128 in / b128 out)
// ---------------------------------------------------------------------------
__global__ void cvt_f32_to_f16_kernel(const float* __restrict__ src,
                                      f16* __restrict__ dst, int n8) {
  int i = blockIdx.x * blockDim.x + threadIdx.x;
  if (i >= n8) return;
  const float4* s = (const float4*)src;
  float4 a = s[2 * i], b = s[2 * i + 1];
  v8h o = {(f16)a.x, (f16)a.y, (f16)a.z, (f16)a.w,
           (f16)b.x, (f16)b.y, (f16)b.z, (f16)b.w};
  *((v8h*)dst + i) = o;
}

// ---------------------------------------------------------------------------
// Kernel 2: tiled WMMA GEMM.  D[M][N] = A[M][K] * B[N][K]^T  (+ bias)
// A row-major K-contiguous, B row-major K-contiguous (= w_qkv / w_y as given).
// Block = 256 threads (8 waves), tile 128x128, BK=32, double-buffered LDS
// filled by ASYNCcnt-tracked global_load_async_to_lds_b128.
// Wave grid 2(M) x 4(N): each wave owns 64x32 = 4x2 WMMA tiles.
// ---------------------------------------------------------------------------
template <bool OUT_F16>
__global__ __launch_bounds__(256) void gemm_f16_kernel(
    const f16* __restrict__ A, const f16* __restrict__ Bm,
    float* __restrict__ Df32, f16* __restrict__ Df16,
    const float* __restrict__ bias, int Nx, int Kx) {
  constexpr int BK = 32, LDT = BK + 8;  // +8 halfs padding, keeps 16B align
  __shared__ f16 As[2][128 * LDT];
  __shared__ f16 Bs[2][128 * LDT];

  const int tid  = threadIdx.x;
  const int lane = tid & 31;
  const int wave = tid >> 5;
  const int wm   = (wave & 1) * 64;   // wave row offset in tile
  const int wn   = (wave >> 1) * 32;  // wave col offset in tile
  const int m0   = blockIdx.y * 128;
  const int n0   = blockIdx.x * 128;
  const int lr   = lane & 15;   // lane % 16
  const int lh   = lane >> 4;   // lane / 16

  v8f acc[4][2];
#pragma unroll
  for (int i = 0; i < 4; ++i)
#pragma unroll
    for (int j = 0; j < 2; ++j) acc[i][j] = zero_v8f();

  // global->LDS mapping: 128 rows x 32 halfs = 512 x 16B chunks, 2 per thread
  const int r0 = tid >> 2;            // rows 0..63
  const int c0 = (tid & 3) * 8;       // half offset 0/8/16/24
  const int r1 = r0 + 64;             // rows 64..127

  auto fill = [&](int buf, int kk) {
    cp16_to_lds(&A[(size_t)(m0 + r0) * Kx + kk + c0], &As[buf][r0 * LDT + c0]);
    cp16_to_lds(&A[(size_t)(m0 + r1) * Kx + kk + c0], &As[buf][r1 * LDT + c0]);
    cp16_to_lds(&Bm[(size_t)(n0 + r0) * Kx + kk + c0], &Bs[buf][r0 * LDT + c0]);
    cp16_to_lds(&Bm[(size_t)(n0 + r1) * Kx + kk + c0], &Bs[buf][r1 * LDT + c0]);
  };

  const int nk = Kx / BK;
  fill(0, 0);
  for (int it = 0; it < nk; ++it) {
    const int cur = it & 1;
    wait_async_lds();      // my fills for buf[cur] complete (ASYNCcnt==0)
    __syncthreads();       // everyone's fills complete; last readers drained
    if (it + 1 < nk) fill(cur ^ 1, (it + 1) * BK);  // overlap with compute

    v16h af[4];
#pragma unroll
    for (int i = 0; i < 4; ++i) {
      const f16* p = &As[cur][(wm + i * 16 + lr) * LDT + lh * 16];
      af[i] = ld2x8(p, p + 8);  // K 0-15 (lanes<16), 16-31 (>=16)
    }
#pragma unroll
    for (int j = 0; j < 2; ++j) {
      v16h bf = ld32B(&Bs[cur][(wn + j * 16 + lr) * LDT + lh * 16]);
#pragma unroll
      for (int i = 0; i < 4; ++i) acc[i][j] = wmma_f16(af[i], bf, acc[i][j]);
    }
    __syncthreads();
  }

  // epilogue: C/D layout -> memory. lane<16: M=r, N=lane; lane>=16: M=r+8.
#pragma unroll
  for (int i = 0; i < 4; ++i)
#pragma unroll
    for (int j = 0; j < 2; ++j) {
      const int col = n0 + wn + j * 16 + lr;
      float bv = 0.0f;
      if (!OUT_F16 && bias) bv = bias[col];
#pragma unroll
      for (int r = 0; r < 8; ++r) {
        const int row = m0 + wm + i * 16 + r + 8 * lh;
        if (OUT_F16)
          Df16[(size_t)row * Nx + col] = (f16)acc[i][j][r];
        else
          Df32[(size_t)row * Nx + col] = acc[i][j][r] + bv;
      }
    }
}

// ---------------------------------------------------------------------------
// Kernel 3: per-head RMS-norm + RoPE.  One wave per (b,l,h); lane owns 4 ch.
// qkv_h: f16 [B*L][3*D].  Writes Q,K f16 [b,h,l,c]; V transposed f16 [b,h,c,l].
// ---------------------------------------------------------------------------
__global__ __launch_bounds__(128) void normrope_kernel(
    const f16* __restrict__ qkv, const float* __restrict__ theta,
    f16* __restrict__ Q, f16* __restrict__ K, f16* __restrict__ Vt) {
  const int lane = threadIdx.x & 31;
  const int w    = blockIdx.x * 4 + (threadIdx.x >> 5);  // (b,l,h) flat
  const int h    = w % H_;
  const int l    = (w / H_) % L_;
  const int b    = w / (H_ * L_);
  const size_t mrow = (size_t)(b * L_ + l) * N3;
  const int c = lane * 4;

  v4h q4 = *(const v4h*)&qkv[mrow + h * C_ + c];
  v4h k4 = *(const v4h*)&qkv[mrow + D_ + h * C_ + c];
  v4h v4 = *(const v4h*)&qkv[mrow + 2 * D_ + h * C_ + c];

  float qf[4], kf[4];
  float sq = 0.0f, sk = 0.0f;
#pragma unroll
  for (int i = 0; i < 4; ++i) {
    qf[i] = (float)q4[i];
    kf[i] = (float)k4[i];
    sq += qf[i] * qf[i];
    sk += kf[i] * kf[i];
  }
#pragma unroll
  for (int msk = 1; msk < 32; msk <<= 1) {
    sq += __shfl_xor(sq, msk, 32);
    sk += __shfl_xor(sk, msk, 32);
  }
  const float rq = rsqrtf(sq * (1.0f / C_) + 1e-5f);
  const float rk = rsqrtf(sk * (1.0f / C_) + 1e-5f);

  const float2 th = *(const float2*)&theta[(size_t)(b * L_ + l) * (D_ / 2) +
                                           h * (C_ / 2) + lane * 2];
  const float c0 = __cosf(th.x), s0 = __sinf(th.x);
  const float c1 = __cosf(th.y), s1 = __sinf(th.y);

  const float q0 = qf[0] * rq, q1 = qf[1] * rq, q2 = qf[2] * rq, q3 = qf[3] * rq;
  const float k0 = kf[0] * rk, k1 = kf[1] * rk, k2 = kf[2] * rk, k3 = kf[3] * rk;

  v4h qo = {(f16)(q0 * c0 - q1 * s0), (f16)(q0 * s0 + q1 * c0),
            (f16)(q2 * c1 - q3 * s1), (f16)(q2 * s1 + q3 * c1)};
  v4h ko = {(f16)(k0 * c0 - k1 * s0), (f16)(k0 * s0 + k1 * c0),
            (f16)(k2 * c1 - k3 * s1), (f16)(k2 * s1 + k3 * c1)};

  const size_t qkbase = ((size_t)(b * H_ + h) * L_ + l) * C_ + c;
  *(v4h*)&Q[qkbase] = qo;
  *(v4h*)&K[qkbase] = ko;

  const size_t vtb = ((size_t)(b * H_ + h) * C_ + c) * L_ + l;
  Vt[vtb]          = v4[0];
  Vt[vtb + L_]     = v4[1];
  Vt[vtb + 2 * L_] = v4[2];
  Vt[vtb + 3 * L_] = v4[3];
}

// ---------------------------------------------------------------------------
// Kernel 4: flash attention.  Block = 4 waves x 16 q-rows; kv tiles of 32.
// Q,K f16 [b,h,l,c]; Vt f16 [b,h,c,l]; Y f16 [b,l, h*C+c].
// Everything is L2-resident (48 MB f16 QKV << 192 MB L2).
// ---------------------------------------------------------------------------
__global__ __launch_bounds__(128) void attn_kernel(
    const f16* __restrict__ Q, const f16* __restrict__ K,
    const f16* __restrict__ Vt, f16* __restrict__ Y) {
  constexpr int PLD = 40;                 // P row stride (halfs), 16B aligned
  __shared__ f16 Plds[4][16 * PLD];       // per-wave P staging (C/D -> A layout)

  const int lane = threadIdx.x & 31;
  const int wave = threadIdx.x >> 5;
  const int q0   = blockIdx.x * 64 + wave * 16;
  const int h    = blockIdx.y;
  const int b    = blockIdx.z;
  const int lr   = lane & 15;
  const int lh   = lane >> 4;

  const size_t bh = (size_t)(b * H_ + h);
  const f16* Qp = Q + bh * L_ * C_;
  const f16* Kp = K + bh * L_ * C_;
  const f16* Vp = Vt + bh * C_ * L_;

  // Q A-fragments (held in registers for the whole kv sweep): 4 k-steps of 32
  v16h qf[4];
#pragma unroll
  for (int s = 0; s < 4; ++s) {
    const f16* p = Qp + (size_t)(q0 + lr) * C_ + s * 32 + lh * 16;
    qf[s] = ld2x8(p, p + 8);
  }

  float mrun[8], lrun[8];
  v8f O[8];
#pragma unroll
  for (int r = 0; r < 8; ++r) { mrun[r] = -1e30f; lrun[r] = 0.0f; }
#pragma unroll
  for (int t = 0; t < 8; ++t) O[t] = zero_v8f();

  f16* Pw = &Plds[wave][0];
  const float sc = 0.08838834764831845f;  // 1/sqrt(128)

  for (int kv0 = 0; kv0 < L_; kv0 += 32) {
    // prefetch next kv tile (global_prefetch_b8) while this tile computes
    if (kv0 + 32 < L_) {
      __builtin_prefetch(Kp + (size_t)(kv0 + 32 + lane) * C_, 0, 0);
      __builtin_prefetch(Vp + (size_t)(lane * 4) * L_ + kv0 + 32, 0, 0);
    }

    // --- S = Q * K^T for a 16x32 tile (two 16x16 WMMA outputs) ---
    v8f S0 = zero_v8f(), S1 = zero_v8f();
#pragma unroll
    for (int s = 0; s < 4; ++s) {
      v16h b0 = ld32B(Kp + (size_t)(kv0 + lr) * C_ + s * 32 + lh * 16);
      v16h b1 = ld32B(Kp + (size_t)(kv0 + 16 + lr) * C_ + s * 32 + lh * 16);
      S0 = wmma_f16(qf[s], b0, S0);
      S1 = wmma_f16(qf[s], b1, S1);
    }

    // --- online softmax; row m lives in VGPR (m%8) of 16-lane half (m/8) ---
#pragma unroll
    for (int r = 0; r < 8; ++r) {
      S0[r] *= sc;
      S1[r] *= sc;
      float mx = fmaxf(S0[r], S1[r]);
#pragma unroll
      for (int msk = 1; msk < 16; msk <<= 1)
        mx = fmaxf(mx, __shfl_xor(mx, msk, 32));
      const float mnew  = fmaxf(mrun[r], mx);
      const float alpha = __expf(mrun[r] - mnew);
      const float p0 = __expf(S0[r] - mnew);
      const float p1 = __expf(S1[r] - mnew);
      float rs = p0 + p1;
#pragma unroll
      for (int msk = 1; msk < 16; msk <<= 1) rs += __shfl_xor(rs, msk, 32);
      lrun[r] = lrun[r] * alpha + rs;
      mrun[r] = mnew;
      S0[r] = p0;
      S1[r] = p1;
#pragma unroll
      for (int t = 0; t < 8; ++t) O[t][r] *= alpha;  // rescale accumulators
    }

    // --- restripe P: C/D layout -> A-operand layout via per-wave LDS ---
#pragma unroll
    for (int r = 0; r < 8; ++r) {
      const int m = r + 8 * lh;
      Pw[m * PLD + lr]      = (f16)S0[r];
      Pw[m * PLD + 16 + lr] = (f16)S1[r];
    }
    v16h pf = ld2x8(&Pw[lr * PLD + lh * 16], &Pw[lr * PLD + lh * 16 + 8]);

    // --- O += P(16x32) * V(32x128): one K=32 WMMA per 16-col c-tile ---
#pragma unroll
    for (int t = 0; t < 8; ++t) {
      v16h vf = ld32B(Vp + (size_t)(t * 16 + lr) * L_ + kv0 + lh * 16);
      O[t] = wmma_f16(pf, vf, O[t]);
    }
  }

  // --- normalize and store Y ---
#pragma unroll
  for (int r = 0; r < 8; ++r) {
    const float inv = 1.0f / lrun[r];
    const int row = q0 + r + 8 * lh;
    const size_t ybase = ((size_t)(b * L_ + row)) * D_ + h * C_;
#pragma unroll
    for (int t = 0; t < 8; ++t)
      Y[ybase + t * 16 + lr] = (f16)(O[t][r] * inv);
  }
}

// ---------------------------------------------------------------------------
// launch
// ---------------------------------------------------------------------------
extern "C" void kernel_launch(void* const* d_in, const int* in_sizes, int n_in,
                              void* d_out, int out_size, void* d_ws,
                              size_t ws_size, hipStream_t stream) {
  (void)in_sizes; (void)n_in; (void)out_size; (void)ws_size;
  const float* x     = (const float*)d_in[0];
  const float* theta = (const float*)d_in[1];
  const float* w_qkv = (const float*)d_in[2];
  const float* w_y   = (const float*)d_in[3];
  const float* b_y   = (const float*)d_in[4];
  float* out = (float*)d_out;

  char* ws = (char*)d_ws;
  f16* x_h    = (f16*)(ws + (size_t)0);            // 16 MB: [M][D]
  f16* wqkv_h = (f16*)(ws + (size_t)(16 << 20));   // 24 MB: [3D][D]
  f16* wy_h   = (f16*)(ws + (size_t)(40 << 20));   //  8 MB: [D][D]
  f16* qkv_h  = (f16*)(ws + (size_t)(48 << 20));   // 48 MB: [M][3D]
  f16* Qh     = (f16*)(ws + (size_t)(96 << 20));   // 16 MB: [b,h,l,c]
  f16* Kh     = (f16*)(ws + (size_t)(112 << 20));  // 16 MB: [b,h,l,c]
  f16* Vt     = (f16*)(ws + (size_t)(128 << 20));  // 16 MB: [b,h,c,l]
  f16* Yh     = (f16*)(ws + (size_t)(144 << 20));  // 16 MB: [M][D]

  // 1) f32 -> f16 conversions
  {
    int n8 = (M_ * D_) / 8;
    cvt_f32_to_f16_kernel<<<(n8 + 255) / 256, 256, 0, stream>>>(x, x_h, n8);
    n8 = (N3 * D_) / 8;
    cvt_f32_to_f16_kernel<<<(n8 + 255) / 256, 256, 0, stream>>>(w_qkv, wqkv_h, n8);
    n8 = (D_ * D_) / 8;
    cvt_f32_to_f16_kernel<<<(n8 + 255) / 256, 256, 0, stream>>>(w_y, wy_h, n8);
  }

  // 2) QKV projection: [4096,2048] x [6144,2048]^T -> f16 [4096,6144]
  gemm_f16_kernel<true><<<dim3(N3 / 128, M_ / 128), 256, 0, stream>>>(
      x_h, wqkv_h, nullptr, qkv_h, nullptr, N3, D_);

  // 3) RMS-norm + RoPE, scatter into Q/K/[b,h,l,c] and V transposed
  normrope_kernel<<<(B_ * L_ * H_) / 4, 128, 0, stream>>>(qkv_h, theta, Qh, Kh, Vt);

  // 4) flash attention -> Y f16 [4096,2048]
  attn_kernel<<<dim3(L_ / 64, H_, B_), 128, 0, stream>>>(Qh, Kh, Vt, Yh);

  // 5) output projection + bias: [4096,2048] x [2048,2048]^T -> f32 out
  gemm_f16_kernel<false><<<dim3(D_ / 128, M_ / 128), 256, 0, stream>>>(
      Yh, wy_h, out, nullptr, b_y, D_, D_);
}
